// AGGemmIntraNode_15496242004358
// MI455X (gfx1250) — matrix operand order repaired and checked by reading, not compiled, not run on hardware
//
#include <hip/hip_runtime.h>

// C[8192,1024] = A[8192,4096] @ B[1024,4096]^T, fp32 in/out.
// Two-pass bf16x3 split-precision GEMM:
//   pass 1: split fp32 -> (hi, lo) bf16 planes in d_ws  (one-time conversion)
//   pass 2: bf16 WMMA GEMM (V_WMMA_F32_16X16X32_BF16), fp32 accumulation.
//           Double-buffered LDS fed by GLOBAL_LOAD_ASYNC_TO_LDS_B128, with
//           s_wait_asynccnt 8 retiring only the previous tile's loads so the
//           next tile's global latency overlaps the current tile's WMMAs.
// Fallback: fused single-pass kernel if ws_size is too small.

typedef __attribute__((ext_vector_type(16))) __bf16 v16bf;
typedef __attribute__((ext_vector_type(8)))  float  v8f;

constexpr int Mtot = 8192;
constexpr int Ktot = 4096;
constexpr int Ntot = 1024;

constexpr int BM = 128;
constexpr int BN = 128;
constexpr int BK = 32;
constexpr int LDK = 40;   // padded LDS row stride (80 B) -> conflict-free b128 frag reads

#if __has_builtin(__builtin_amdgcn_global_load_async_to_lds_b128)
#define HAVE_ASYNC_LDS 1
#else
#define HAVE_ASYNC_LDS 0
#endif

// Builtin prototype (from hipcc diagnostic): arg0 "int __vector(4) __device__ *"
// (non-const v4i pointer in addrspace(1)); arg1 the LDS-side v4i pointer.
typedef int v4i_vs __attribute__((vector_size(16)));
typedef __attribute__((address_space(1))) v4i_vs* as1_v4i;
typedef __attribute__((address_space(3))) v4i_vs* as3_v4i;

union Frag {
    uint4 q[2];
    v16bf v;
};

__device__ __forceinline__ unsigned pack_bf16(__bf16 a, __bf16 b) {
    unsigned lo = (unsigned)__builtin_bit_cast(unsigned short, a);
    unsigned hi = (unsigned)__builtin_bit_cast(unsigned short, b);
    return (hi << 16) | lo;
}

__device__ __forceinline__ void split4(const float4 f,
                                       unsigned& h01, unsigned& h23,
                                       unsigned& l01, unsigned& l23) {
    __bf16 h0 = (__bf16)f.x, h1 = (__bf16)f.y, h2 = (__bf16)f.z, h3 = (__bf16)f.w;
    __bf16 l0 = (__bf16)(f.x - (float)h0);
    __bf16 l1 = (__bf16)(f.y - (float)h1);
    __bf16 l2 = (__bf16)(f.z - (float)h2);
    __bf16 l3 = (__bf16)(f.w - (float)h3);
    h01 = pack_bf16(h0, h1);
    h23 = pack_bf16(h2, h3);
    l01 = pack_bf16(l0, l1);
    l23 = pack_bf16(l2, l3);
}

// ---------------- pass 1: one-time fp32 -> bf16 hi/lo split ----------------
__global__ __launch_bounds__(256)
void split_bf16_kernel(const float* __restrict__ X,
                       __bf16* __restrict__ Hi, __bf16* __restrict__ Lo,
                       long n4) {
    long i = (long)blockIdx.x * blockDim.x + threadIdx.x;
    const long stride = (long)gridDim.x * blockDim.x;
    for (; i < n4; i += stride) {
        const float4 f = ((const float4*)X)[i];
        unsigned h01, h23, l01, l23;
        split4(f, h01, h23, l01, l23);
        ((uint2*)Hi)[i] = make_uint2(h01, h23);
        ((uint2*)Lo)[i] = make_uint2(l01, l23);
    }
}

// ---------------- shared fragment loaders (ISA 7.12.2 layouts) ----------------
// A operand 16x32 bf16: lane row M=lane&15; lanes 0-15 take K 0..7 & 16..23,
// lanes 16-31 take K 8..15 & 24..31 -> two contiguous 16B reads.
__device__ __forceinline__ void load_fragA(const __bf16* base, int row, int kb, Frag& f) {
    const __bf16* p = base + row * LDK + kb;
    f.q[0] = *(const uint4*)(p);
    f.q[1] = *(const uint4*)(p + 16);
}
// B operand 32x16 bf16: lane col N=lane&15; lanes 0-15 K 0..7 & 16..23,
// lanes 16-31 K 8..15 & 24..31 -> two contiguous 16B reads.
__device__ __forceinline__ void load_fragB(const __bf16* base, int col, int kb, Frag& f) {
    const __bf16* p = base + col * LDK + kb;
    f.q[0] = *(const uint4*)(p);
    f.q[1] = *(const uint4*)(p + 16);
}

__device__ __forceinline__ void cp16_g2l(const __bf16* g, __bf16* l) {
#if HAVE_ASYNC_LDS
    __builtin_amdgcn_global_load_async_to_lds_b128((as1_v4i)g, (as3_v4i)l, 0, 0);
#else
    *(uint4*)l = *(const uint4*)g;
#endif
}

template <int N>
__device__ __forceinline__ void wait_async_le() {
#if HAVE_ASYNC_LDS
#if __has_builtin(__builtin_amdgcn_s_wait_asynccnt)
    __builtin_amdgcn_s_wait_asynccnt(N);
#else
    asm volatile("s_wait_asynccnt %0" ::"i"(N) : "memory");
#endif
#endif
}

// Issue one BKxBM/BN tile (all 4 planes) as async b128 copies: 8 per thread.
__device__ __forceinline__ void issue_tile(const __bf16* __restrict__ Ahi,
                                           const __bf16* __restrict__ Alo,
                                           const __bf16* __restrict__ Bhi,
                                           const __bf16* __restrict__ Blo,
                                           __bf16* dAhi, __bf16* dAlo,
                                           __bf16* dBhi, __bf16* dBlo,
                                           long blockM, long blockN, int kt, int c0) {
#pragma unroll
    for (int c = c0; c < c0 + 2; ++c) {
        const int row  = c >> 2;
        const int koff = (c & 3) * 8;
        const long gA = (blockM + row) * (long)Ktot + kt + koff;
        const long gB = (blockN + row) * (long)Ktot + kt + koff;
        const int  l  = row * LDK + koff;
        cp16_g2l(Ahi + gA, dAhi + l);
        cp16_g2l(Alo + gA, dAlo + l);
        cp16_g2l(Bhi + gB, dBhi + l);
        cp16_g2l(Blo + gB, dBlo + l);
    }
}

// ---------------- pass 2: bf16x3 GEMM on pre-split planes ----------------
__global__ __launch_bounds__(256)
void ag_gemm_presplit_kernel(const __bf16* __restrict__ Ahi, const __bf16* __restrict__ Alo,
                             const __bf16* __restrict__ Bhi, const __bf16* __restrict__ Blo,
                             float* __restrict__ C) {
    // double-buffered tile planes: 2 x 4 x 10240 B = 80 KB (CDNA5: up to 320 KB/WG)
    __shared__ __bf16 sAhi[2][BM * LDK];
    __shared__ __bf16 sAlo[2][BM * LDK];
    __shared__ __bf16 sBhi[2][BN * LDK];
    __shared__ __bf16 sBlo[2][BN * LDK];

    const int tid  = threadIdx.x;
    const int lane = tid & 31;
    const int wave = tid >> 5;
    const int wm   = wave >> 2;      // 0..1 -> 64 output rows per wave
    const int wn   = wave & 3;       // 0..3 -> 32 output cols per wave

    const long blockM = (long)blockIdx.y * BM;
    const long blockN = (long)blockIdx.x * BN;

    v8f acc[4][2];
#pragma unroll
    for (int m = 0; m < 4; ++m)
#pragma unroll
        for (int n = 0; n < 2; ++n) {
            v8f z = {0.f, 0.f, 0.f, 0.f, 0.f, 0.f, 0.f, 0.f};
            acc[m][n] = z;
        }

    const int c0   = tid * 2;        // 2 b128 chunks per plane per thread
    const int arow = lane & 15;
    const int kbA  = (lane >> 4) * 8;
    const int kbB  = (lane >> 4) * 8;

    // prologue: prefetch tile 0 into buffer 0
    issue_tile(Ahi, Alo, Bhi, Blo,
               sAhi[0], sAlo[0], sBhi[0], sBlo[0],
               blockM, blockN, 0, c0);

    for (int kt = 0; kt < Ktot; kt += BK) {
        const int cur = (kt >> 5) & 1;

        if (kt + BK < Ktot) {
            // prefetch next tile into the other buffer, then retire only the
            // previous 8 async loads (per thread): asynccnt <= 8.
            issue_tile(Ahi, Alo, Bhi, Blo,
                       sAhi[cur ^ 1], sAlo[cur ^ 1], sBhi[cur ^ 1], sBlo[cur ^ 1],
                       blockM, blockN, kt + BK, c0);
            wait_async_le<8>();
        } else {
            wait_async_le<0>();
        }
        __syncthreads();   // tile kt visible to all waves

        const __bf16* a_hi = sAhi[cur];
        const __bf16* a_lo = sAlo[cur];
        const __bf16* b_hi = sBhi[cur];
        const __bf16* b_lo = sBlo[cur];

        Frag bh[2], bl[2];
#pragma unroll
        for (int n = 0; n < 2; ++n) {
            load_fragB(b_hi, wn * 32 + n * 16 + arow, kbB, bh[n]);
            load_fragB(b_lo, wn * 32 + n * 16 + arow, kbB, bl[n]);
        }

#pragma unroll
        for (int m = 0; m < 4; ++m) {
            Frag ah, al;
            load_fragA(a_hi, wm * 64 + m * 16 + arow, kbA, ah);
            load_fragA(a_lo, wm * 64 + m * 16 + arow, kbA, al);
#pragma unroll
            for (int n = 0; n < 2; ++n) {
                // hi*hi + hi*lo + lo*hi  (lo*lo dropped, ~2^-16 relative)
                acc[m][n] = __builtin_amdgcn_wmma_f32_16x16x32_bf16(
                    false, ah.v, false, bh[n].v, (short)0, acc[m][n], false, false);
                acc[m][n] = __builtin_amdgcn_wmma_f32_16x16x32_bf16(
                    false, ah.v, false, bl[n].v, (short)0, acc[m][n], false, false);
                acc[m][n] = __builtin_amdgcn_wmma_f32_16x16x32_bf16(
                    false, al.v, false, bh[n].v, (short)0, acc[m][n], false, false);
            }
        }
        __syncthreads();   // all waves done reading buf[cur] before it is re-filled
    }

    // C/D layout: lane holds col N=lane&15; VGPR r holds row M = r + 8*(lane>>4).
    const int cn    = (int)blockN + wn * 32 + (lane & 15);
    const int rhalf = (lane >> 4) * 8;
#pragma unroll
    for (int m = 0; m < 4; ++m) {
        const long rbase = blockM + wm * 64 + m * 16 + rhalf;
#pragma unroll
        for (int n = 0; n < 2; ++n) {
#pragma unroll
            for (int r = 0; r < 8; ++r) {
                C[(rbase + r) * (long)Ntot + cn + n * 16] = acc[m][n][r];
            }
        }
    }
}

// ---------------- fallback: fused single-pass kernel ----------------
__global__ __launch_bounds__(256)
void ag_gemm_fused_kernel(const float* __restrict__ A,
                          const float* __restrict__ B,
                          float* __restrict__ C) {
    __shared__ __bf16 sAhi[BM * LDK];
    __shared__ __bf16 sAlo[BM * LDK];
    __shared__ __bf16 sBhi[BN * LDK];
    __shared__ __bf16 sBlo[BN * LDK];

    const int tid  = threadIdx.x;
    const int lane = tid & 31;
    const int wave = tid >> 5;
    const int wm   = wave >> 2;
    const int wn   = wave & 3;

    const long blockM = (long)blockIdx.y * BM;
    const long blockN = (long)blockIdx.x * BN;

    v8f acc[4][2];
#pragma unroll
    for (int m = 0; m < 4; ++m)
#pragma unroll
        for (int n = 0; n < 2; ++n) {
            v8f z = {0.f, 0.f, 0.f, 0.f, 0.f, 0.f, 0.f, 0.f};
            acc[m][n] = z;
        }

    const int lr = tid >> 3;
    const int lc = (tid & 7) * 4;

    const int arow = lane & 15;
    const int kbA  = (lane >> 4) * 8;
    const int kbB  = (lane >> 4) * 8;

    for (int kt = 0; kt < Ktot; kt += BK) {
#pragma unroll
        for (int i = 0; i < 4; ++i) {
            const int row = lr + i * 32;
            const float4 fa = *(const float4*)(A + (blockM + row) * (long)Ktot + kt + lc);
            const float4 fb = *(const float4*)(B + (blockN + row) * (long)Ktot + kt + lc);
            unsigned h01, h23, l01, l23;
            split4(fa, h01, h23, l01, l23);
            *(uint2*)(sAhi + row * LDK + lc) = make_uint2(h01, h23);
            *(uint2*)(sAlo + row * LDK + lc) = make_uint2(l01, l23);
            split4(fb, h01, h23, l01, l23);
            *(uint2*)(sBhi + row * LDK + lc) = make_uint2(h01, h23);
            *(uint2*)(sBlo + row * LDK + lc) = make_uint2(l01, l23);
        }
        __syncthreads();

        Frag bh[2], bl[2];
#pragma unroll
        for (int n = 0; n < 2; ++n) {
            load_fragB(sBhi, wn * 32 + n * 16 + arow, kbB, bh[n]);
            load_fragB(sBlo, wn * 32 + n * 16 + arow, kbB, bl[n]);
        }
#pragma unroll
        for (int m = 0; m < 4; ++m) {
            Frag ah, al;
            load_fragA(sAhi, wm * 64 + m * 16 + arow, kbA, ah);
            load_fragA(sAlo, wm * 64 + m * 16 + arow, kbA, al);
#pragma unroll
            for (int n = 0; n < 2; ++n) {
                acc[m][n] = __builtin_amdgcn_wmma_f32_16x16x32_bf16(
                    false, ah.v, false, bh[n].v, (short)0, acc[m][n], false, false);
                acc[m][n] = __builtin_amdgcn_wmma_f32_16x16x32_bf16(
                    false, ah.v, false, bl[n].v, (short)0, acc[m][n], false, false);
                acc[m][n] = __builtin_amdgcn_wmma_f32_16x16x32_bf16(
                    false, al.v, false, bh[n].v, (short)0, acc[m][n], false, false);
            }
        }
        __syncthreads();
    }

    const int cn    = (int)blockN + wn * 32 + (lane & 15);
    const int rhalf = (lane >> 4) * 8;
#pragma unroll
    for (int m = 0; m < 4; ++m) {
        const long rbase = blockM + wm * 64 + m * 16 + rhalf;
#pragma unroll
        for (int n = 0; n < 2; ++n) {
#pragma unroll
            for (int r = 0; r < 8; ++r) {
                C[(rbase + r) * (long)Ntot + cn + n * 16] = acc[m][n][r];
            }
        }
    }
}

extern "C" void kernel_launch(void* const* d_in, const int* in_sizes, int n_in,
                              void* d_out, int out_size, void* d_ws, size_t ws_size,
                              hipStream_t stream) {
    const float* A = (const float*)d_in[0];   // [8,1024,4096] == [8192,4096] contiguous
    const float* B = (const float*)d_in[1];   // [1024,4096]
    float* C = (float*)d_out;                 // [8192,1024]

    const size_t aElems = (size_t)Mtot * Ktot;
    const size_t bElems = (size_t)Ntot * Ktot;
    const size_t need   = 2 * (aElems + bElems) * sizeof(unsigned short);  // 144 MiB

    dim3 grid(Ntot / BN, Mtot / BM);          // 8 x 64 = 512 workgroups
    dim3 block(256);                          // 8 waves (wave32)

    if (ws_size >= need) {
        __bf16* Ahi = (__bf16*)d_ws;
        __bf16* Alo = Ahi + aElems;
        __bf16* Bhi = Alo + aElems;
        __bf16* Blo = Bhi + bElems;

        split_bf16_kernel<<<8192, 256, 0, stream>>>(A, Ahi, Alo, (long)(aElems / 4));
        split_bf16_kernel<<<1024, 256, 0, stream>>>(B, Bhi, Blo, (long)(bElems / 4));
        ag_gemm_presplit_kernel<<<grid, block, 0, stream>>>(Ahi, Alo, Bhi, Blo, C);
    } else {
        ag_gemm_fused_kernel<<<grid, block, 0, stream>>>(A, B, C);
    }
}
